// LongcatMoe_88235808129201
// MI455X (gfx1250) — matrix operand order repaired and checked by reading, not compiled, not run on hardware
//
#include <hip/hip_runtime.h>
#include <hip/hip_bf16.h>

// LongCat MoE for MI455X (gfx1250): f32 router + top-2, bf16 WMMA expert GEMMs,
// fused SiLU, atomic-f32 combine scatter. Sparse compute (~39 GFLOP) via
// per-expert token gather. BM=32 tiles (2x B-fragment reuse) with a ping-pong
// two-stage software pipeline (no register rotation -> no v_mov chains, no
// WMMA->VALU hazard NOPs; loads lead their WMMAs by a full stage).

constexpr int T = 4096;
constexpr int H = 1024;
constexpr int I = 768;
constexpr int E = 8;

typedef __bf16 bf16_t;
typedef __attribute__((ext_vector_type(16))) __bf16 v16bf;
typedef __attribute__((ext_vector_type(8)))  __bf16 v8bf;
typedef __attribute__((ext_vector_type(8)))  float  v8f;

__device__ __forceinline__ v8f vzero8f() {
  v8f z;
#pragma unroll
  for (int i = 0; i < 8; ++i) z[i] = 0.f;
  return z;
}

// A-fragment (16x32 bf16, M x K). Lanes 0-15 hold row M=lane,
// K in {k0..k0+15}; lanes 16-31: K in {k0+16..k0+31}.
// lanebase = smem_tile + (lane&15)*ldk + ((lane>>4)<<4)
__device__ __forceinline__ v16bf load_a(const bf16_t* lanebase, int k0) {
  v8bf lo = *(const v8bf*)(lanebase + k0);
  v8bf hi = *(const v8bf*)(lanebase + k0 + 8);
  v16bf r;
#pragma unroll
  for (int i = 0; i < 8; ++i) { r[i] = lo[i]; r[8 + i] = hi[i]; }
  return r;
}

// B-fragment (32x16 bf16, K x N). Column n = lane&15 is a K-contiguous weight
// row; lanes 0-15 take K k0..k0+15, lanes 16-31 take K k0+16..k0+31.
// lanebase = mat + (lane&15)*ldk + ((lane>>4)<<4)
__device__ __forceinline__ v16bf load_b(const bf16_t* lanebase, int k0) {
  return *(const v16bf*)(lanebase + k0);
}

__device__ __forceinline__ v8f wmma_bf16(v16bf a, v16bf b, v8f c) {
  return __builtin_amdgcn_wmma_f32_16x16x32_bf16(false, a, false, b, (short)0, c, false, false);
}

// ---------------- f32 -> bf16 conversion (8 elems / thread) ----------------
__global__ void cvt_f32_bf16(const float* __restrict__ src, bf16_t* __restrict__ dst, int n8) {
  int i = blockIdx.x * blockDim.x + threadIdx.x;
  if (i >= n8) return;
  const float4* s = (const float4*)src + (size_t)i * 2;
  float4 a = s[0], b = s[1];
  v8bf o;
  o[0] = (bf16_t)a.x; o[1] = (bf16_t)a.y; o[2] = (bf16_t)a.z; o[3] = (bf16_t)a.w;
  o[4] = (bf16_t)b.x; o[5] = (bf16_t)b.y; o[6] = (bf16_t)b.z; o[7] = (bf16_t)b.w;
  *((v8bf*)dst + i) = o;
}

// ---------------- router: logits, softmax, top-2, token lists ----------------
__global__ void __launch_bounds__(256) router_kernel(
    const float* __restrict__ x, const float* __restrict__ gw,
    float* __restrict__ combine, int* __restrict__ counts, int* __restrict__ tidx) {
  __shared__ float sgw[E * H];  // 32 KB gate weights
  for (int i = threadIdx.x; i < E * H / 4; i += blockDim.x)
    ((float4*)sgw)[i] = ((const float4*)gw)[i];
  __syncthreads();

  int wave = threadIdx.x >> 5, lane = threadIdx.x & 31;
  int t = blockIdx.x * 8 + wave;

  float acc[E];
#pragma unroll
  for (int e = 0; e < E; ++e) acc[e] = 0.f;
  const float* xr = x + (size_t)t * H;
  for (int h = lane; h < H; h += 32) {
    float xv = xr[h];
#pragma unroll
    for (int e = 0; e < E; ++e) acc[e] += xv * sgw[e * H + h];
  }
#pragma unroll
  for (int off = 16; off > 0; off >>= 1)
#pragma unroll
    for (int e = 0; e < E; ++e) acc[e] += __shfl_xor(acc[e], off, 32);

  if (lane == 0) {
    float mx = acc[0];
#pragma unroll
    for (int e = 1; e < E; ++e) mx = fmaxf(mx, acc[e]);
    float p[E], s = 0.f;
#pragma unroll
    for (int e = 0; e < E; ++e) { p[e] = __expf(acc[e] - mx); s += p[e]; }
    float inv = 1.f / s;
    int i1 = 0; float v1 = p[0];
#pragma unroll
    for (int e = 1; e < E; ++e) if (p[e] > v1) { v1 = p[e]; i1 = e; }
    int i2 = -1; float v2 = -1.f;
#pragma unroll
    for (int e = 0; e < E; ++e) if (e != i1 && p[e] > v2) { v2 = p[e]; i2 = e; }
#pragma unroll
    for (int e = 0; e < E; ++e)
      combine[(size_t)t * E + e] = (e == i1) ? v1 * inv : ((e == i2) ? v2 * inv : 0.f);
    int p1 = atomicAdd(&counts[i1], 1); tidx[i1 * T + p1] = t;
    int p2 = atomicAdd(&counts[i2], 1); tidx[i2 * T + p2] = t;
  }
}

__global__ void offsets_kernel(const int* __restrict__ counts, int* __restrict__ offs) {
  if (threadIdx.x == 0 && blockIdx.x == 0) {
    int s = 0;
    for (int e = 0; e < E; ++e) { offs[e] = s; s += counts[e]; }
    offs[E] = s;
  }
}

// -------- FFN1: h = x_g @ w13^T, a = silu(g)*u, bf16 activations --------
// grid (3, T/32, E), block 256 (8 waves). Block: 32 gathered rows x 256 a-cols.
// Wave: 2 m-tiles x 2 col-tiles x {gate,up} = 8 WMMA tiles, SiLU fused in regs.
__global__ void __launch_bounds__(256) ffn1_kernel(
    const bf16_t* __restrict__ xb, const bf16_t* __restrict__ w13b,
    const int* __restrict__ counts, const int* __restrict__ offs,
    const int* __restrict__ tidx, bf16_t* __restrict__ act) {
  int e = blockIdx.z;
  int cnt = counts[e];
  int m0 = blockIdx.y * 32;
  if (m0 >= cnt) return;

  __shared__ __align__(16) bf16_t sA[32 * H];  // 64 KB gathered token rows
  int tid = threadIdx.x;
  {
    int r = tid >> 3, c = tid & 7;  // 32 rows x 8 chunks of 128 bf16
    int mrow = m0 + r; if (mrow > cnt - 1) mrow = cnt - 1;
    int token = tidx[e * T + mrow];
    const uint4* src = (const uint4*)(xb + (size_t)token * H + c * 128);
    uint4* dst = (uint4*)(sA + r * H + c * 128);
#pragma unroll
    for (int i = 0; i < 16; ++i) dst[i] = src[i];
  }
  __syncthreads();

  int wave = tid >> 5, lane = tid & 31;
  int n0 = blockIdx.x * 256 + wave * 32;  // a-col base, 0..767 step 32
  const bf16_t* Bw = w13b + (size_t)e * (2 * I) * H;

  const bf16_t* aB[2];
#pragma unroll
  for (int mi = 0; mi < 2; ++mi)
    aB[mi] = sA + (size_t)(mi * 16 + (lane & 15)) * H + ((lane >> 4) << 4);
  const bf16_t* bgB[2];
  const bf16_t* buB[2];
#pragma unroll
  for (int ci = 0; ci < 2; ++ci) {
    bgB[ci] = Bw + (size_t)(n0 + ci * 16 + (lane & 15)) * H + ((lane >> 4) << 4);
    buB[ci] = Bw + (size_t)(I + n0 + ci * 16 + (lane & 15)) * H + ((lane >> 4) << 4);
  }

  v8f accg[2][2], accu[2][2];
#pragma unroll
  for (int mi = 0; mi < 2; ++mi)
#pragma unroll
    for (int ci = 0; ci < 2; ++ci) { accg[mi][ci] = vzero8f(); accu[mi][ci] = vzero8f(); }

  // ping-pong two-stage pipeline over K (stage = 32)
  v16bf aX[2], gX[2], uX[2];  // stage X
  v16bf aY[2], gY[2], uY[2];  // stage Y
  auto loadX = [&](int k) {
#pragma unroll
    for (int mi = 0; mi < 2; ++mi) aX[mi] = load_a(aB[mi], k);
#pragma unroll
    for (int ci = 0; ci < 2; ++ci) { gX[ci] = load_b(bgB[ci], k); uX[ci] = load_b(buB[ci], k); }
  };
  auto loadY = [&](int k) {
#pragma unroll
    for (int mi = 0; mi < 2; ++mi) aY[mi] = load_a(aB[mi], k);
#pragma unroll
    for (int ci = 0; ci < 2; ++ci) { gY[ci] = load_b(bgB[ci], k); uY[ci] = load_b(buB[ci], k); }
  };
  auto mmaX = [&]() {
#pragma unroll
    for (int mi = 0; mi < 2; ++mi)
#pragma unroll
      for (int ci = 0; ci < 2; ++ci) {
        accg[mi][ci] = wmma_bf16(aX[mi], gX[ci], accg[mi][ci]);
        accu[mi][ci] = wmma_bf16(aX[mi], uX[ci], accu[mi][ci]);
      }
  };
  auto mmaY = [&]() {
#pragma unroll
    for (int mi = 0; mi < 2; ++mi)
#pragma unroll
      for (int ci = 0; ci < 2; ++ci) {
        accg[mi][ci] = wmma_bf16(aY[mi], gY[ci], accg[mi][ci]);
        accu[mi][ci] = wmma_bf16(aY[mi], uY[ci], accu[mi][ci]);
      }
  };

  loadX(0);
  int k0 = 0;
  for (; k0 + 128 <= H; k0 += 64) {
    loadY(k0 + 32);
    mmaX();
    loadX(k0 + 64);
    mmaY();
  }
  // exit with X holding data at k0; one Y stage remains
  loadY(k0 + 32);
  mmaX();
  mmaY();

  // epilogue: SiLU fuse + bf16 store to compacted activation rows
  int col = lane & 15;
  int rbase = (lane >> 4) << 3;
  size_t arow0 = (size_t)offs[e] + m0;
#pragma unroll
  for (int mi = 0; mi < 2; ++mi)
#pragma unroll
    for (int ci = 0; ci < 2; ++ci)
#pragma unroll
      for (int r = 0; r < 8; ++r) {
        int m = mi * 16 + rbase + r;
        if (m0 + m < cnt) {
          float g = accg[mi][ci][r], u = accu[mi][ci][r];
          float a_val = g * (1.f / (1.f + __expf(-g))) * u;
          act[(arow0 + m) * I + n0 + ci * 16 + col] = (bf16_t)a_val;
        }
      }
}

// -------- FFN2: o = a @ w2^T, out[t] += combine[t,e] * o (atomic f32) --------
// grid (4, T/32, E), block 256 (8 waves). Block: 32 act rows x 256 of H cols.
__global__ void __launch_bounds__(256) ffn2_kernel(
    const bf16_t* __restrict__ act, const bf16_t* __restrict__ w2b,
    const int* __restrict__ counts, const int* __restrict__ offs,
    const int* __restrict__ tidx, const float* __restrict__ combine,
    float* __restrict__ out) {
  int e = blockIdx.z;
  int cnt = counts[e];
  int m0 = blockIdx.y * 32;
  if (m0 >= cnt) return;

  __shared__ __align__(16) bf16_t sA[32 * I];  // 48 KB activation rows
  int tid = threadIdx.x;
  {
    int r = tid >> 3, c = tid & 7;  // 32 rows x 8 chunks of 96 bf16
    int mrow = m0 + r; if (mrow > cnt - 1) mrow = cnt - 1;
    const uint4* src = (const uint4*)(act + ((size_t)offs[e] + mrow) * I + c * 96);
    uint4* dst = (uint4*)(sA + r * I + c * 96);
#pragma unroll
    for (int i = 0; i < 12; ++i) dst[i] = src[i];
  }
  __syncthreads();

  int wave = tid >> 5, lane = tid & 31;
  int n0 = blockIdx.x * 256 + wave * 32;  // 0..1023 step 32
  const bf16_t* Bw = w2b + (size_t)e * H * I;

  const bf16_t* aB[2];
#pragma unroll
  for (int mi = 0; mi < 2; ++mi)
    aB[mi] = sA + (size_t)(mi * 16 + (lane & 15)) * I + ((lane >> 4) << 4);
  const bf16_t* bB[2];
#pragma unroll
  for (int ci = 0; ci < 2; ++ci)
    bB[ci] = Bw + (size_t)(n0 + ci * 16 + (lane & 15)) * I + ((lane >> 4) << 4);

  v8f acc[2][2];
#pragma unroll
  for (int mi = 0; mi < 2; ++mi)
#pragma unroll
    for (int ci = 0; ci < 2; ++ci) acc[mi][ci] = vzero8f();

  // ping-pong two-stage pipeline over K (stage = 32)
  v16bf aX[2], bX[2], aY[2], bY[2];
  auto loadX = [&](int k) {
#pragma unroll
    for (int mi = 0; mi < 2; ++mi) aX[mi] = load_a(aB[mi], k);
#pragma unroll
    for (int ci = 0; ci < 2; ++ci) bX[ci] = load_b(bB[ci], k);
  };
  auto loadY = [&](int k) {
#pragma unroll
    for (int mi = 0; mi < 2; ++mi) aY[mi] = load_a(aB[mi], k);
#pragma unroll
    for (int ci = 0; ci < 2; ++ci) bY[ci] = load_b(bB[ci], k);
  };
  auto mmaX = [&]() {
#pragma unroll
    for (int mi = 0; mi < 2; ++mi)
#pragma unroll
      for (int ci = 0; ci < 2; ++ci) acc[mi][ci] = wmma_bf16(aX[mi], bX[ci], acc[mi][ci]);
  };
  auto mmaY = [&]() {
#pragma unroll
    for (int mi = 0; mi < 2; ++mi)
#pragma unroll
      for (int ci = 0; ci < 2; ++ci) acc[mi][ci] = wmma_bf16(aY[mi], bY[ci], acc[mi][ci]);
  };

  loadX(0);
  int k0 = 0;
  for (; k0 + 128 <= I; k0 += 64) {
    loadY(k0 + 32);
    mmaX();
    loadX(k0 + 64);
    mmaY();
  }
  loadY(k0 + 32);
  mmaX();
  mmaY();

  // epilogue: scatter-accumulate weighted outputs (2 addends/token -> exact)
  int col = lane & 15;
  int rbase = (lane >> 4) << 3;
  int   tok[2][8];
  float wgt[2][8];
#pragma unroll
  for (int mi = 0; mi < 2; ++mi)
#pragma unroll
    for (int r = 0; r < 8; ++r) {
      int m = mi * 16 + rbase + r;
      if (m0 + m < cnt) {
        int tk = tidx[e * T + m0 + m];
        tok[mi][r] = tk;
        wgt[mi][r] = combine[(size_t)tk * E + e];
      } else { tok[mi][r] = -1; wgt[mi][r] = 0.f; }
    }
#pragma unroll
  for (int mi = 0; mi < 2; ++mi)
#pragma unroll
    for (int ci = 0; ci < 2; ++ci)
#pragma unroll
      for (int r = 0; r < 8; ++r)
        if (tok[mi][r] >= 0)
          atomicAdd(&out[(size_t)tok[mi][r] * H + n0 + ci * 16 + col],
                    wgt[mi][r] * acc[mi][ci][r]);
}

extern "C" void kernel_launch(void* const* d_in, const int* in_sizes, int n_in,
                              void* d_out, int out_size, void* d_ws, size_t ws_size,
                              hipStream_t stream) {
  const float* x   = (const float*)d_in[0];  // (T,H)
  const float* gw  = (const float*)d_in[1];  // (E,H)
  const float* w13 = (const float*)d_in[2];  // (E,2I,H)
  const float* w2  = (const float*)d_in[3];  // (E,H,I)
  float* out = (float*)d_out;                // (T,H) f32

  // ---- workspace carving (256B aligned regions), total ~59 MB ----
  size_t off = 0;
  auto carve = [&](size_t bytes) -> void* {
    void* r = (char*)d_ws + off;
    off += (bytes + 255) & ~(size_t)255;
    return r;
  };
  float*  combine = (float*)carve(sizeof(float) * T * E);
  int*    counts  = (int*)carve(sizeof(int) * E);
  int*    offs    = (int*)carve(sizeof(int) * (E + 1));
  int*    tidx    = (int*)carve(sizeof(int) * E * T);
  bf16_t* xb      = (bf16_t*)carve(sizeof(bf16_t) * (size_t)T * H);
  bf16_t* w13b    = (bf16_t*)carve(sizeof(bf16_t) * (size_t)E * 2 * I * H);
  bf16_t* w2b     = (bf16_t*)carve(sizeof(bf16_t) * (size_t)E * H * I);
  bf16_t* act     = (bf16_t*)carve(sizeof(bf16_t) * (size_t)2 * T * I);  // sum n_e == 2T

  hipMemsetAsync(out, 0, sizeof(float) * (size_t)T * H, stream);
  hipMemsetAsync(counts, 0, sizeof(int) * E, stream);

  router_kernel<<<T / 8, 256, 0, stream>>>(x, gw, combine, counts, tidx);
  offsets_kernel<<<1, 32, 0, stream>>>(counts, offs);

  {
    int n8 = T * H / 8;
    cvt_f32_bf16<<<(n8 + 255) / 256, 256, 0, stream>>>(x, xb, n8);
  }
  {
    int n8 = E * 2 * I * H / 8;
    cvt_f32_bf16<<<(n8 + 255) / 256, 256, 0, stream>>>(w13, w13b, n8);
  }
  {
    int n8 = E * H * I / 8;
    cvt_f32_bf16<<<(n8 + 255) / 256, 256, 0, stream>>>(w2, w2b, n8);
  }

  ffn1_kernel<<<dim3(3, T / 32, E), 256, 0, stream>>>(xb, w13b, counts, offs, tidx, act);
  ffn2_kernel<<<dim3(4, T / 32, E), 256, 0, stream>>>(act, w2b, counts, offs, tidx, combine, out);
}
